// TopKAutoEncoder_60576218743382
// MI455X (gfx1250) — matrix operand order repaired and checked by reading, not compiled, not run on hardware
//
#include <hip/hip_runtime.h>
#include <hip/hip_bf16.h>
#include <stdint.h>

// ---------------------------------------------------------------------------
// TopK sparse autoencoder forward for MI455X (gfx1250, wave32, WMMA + TDM)
//   encode:  post = relu((x - b_dec) @ W_enc^T + b_enc)      [8192 x 16384]
//   topk:    keep top-32 per row, zero the rest              (radix select)
//   decode:  x_rec = acts @ W_dec^T + b_dec  (K-sparse gather of W_enc rows,
//            since W_dec == W_enc^T by construction)
//   loss:    mean_n ||x_rec - x||^2
// GEMM uses Tensor Data Mover double-buffering (global -> LDS f32 stage),
// in-LDS split to bf16 hi/lo, and v_wmma_f32_16x16x32_bf16 (hi*hi+hi*lo+lo*hi).
// ---------------------------------------------------------------------------

typedef __bf16 bf16_t;
typedef __attribute__((ext_vector_type(16))) __bf16 v16bf;
typedef __attribute__((ext_vector_type(8)))  __bf16 v8bf;
typedef __attribute__((ext_vector_type(4)))  __bf16 v4bf;
typedef __attribute__((ext_vector_type(8)))  float  v8f;
typedef __attribute__((ext_vector_type(4)))  unsigned int u32x4;
typedef __attribute__((ext_vector_type(8)))  int i32x8;
typedef __attribute__((ext_vector_type(4)))  int i32x4;

constexpr int NROWS  = 8192;
constexpr int DACT   = 1024;
constexpr int DDICT  = 16384;
constexpr int TOPK   = 32;

// flat d_out layout (floats): loss | x_rec | acts | l2 | l1
constexpr long long OUT_XREC = 1;
constexpr long long OUT_ACTS = 1 + (long long)NROWS * DACT;                 // 8388609
constexpr long long OUT_L2   = OUT_ACTS + (long long)NROWS * DDICT;         // 142606337
constexpr long long OUT_L1   = OUT_L2 + 1;

// ------------------------------ GEMM tiling --------------------------------
constexpr int BM = 128;        // rows of x per block
constexpr int BN = 128;        // dict columns per block
constexpr int BK = 32;         // k-slice (== WMMA K)
constexpr int LDSS = 40;       // bf16 row stride in LDS (32 + 8 pad -> 80B, bank-conflict free)
constexpr int KSTEPS = DACT / BK;   // 32

__device__ inline v16bf frag_cat(v8bf a, v8bf b) {
  return __builtin_shufflevector(a, b, 0,1,2,3,4,5,6,7,8,9,10,11,12,13,14,15);
}

// A-fragment (16x32 bf16, M x K): lane<16 holds M=lane, K {0..7, 16..23};
// lane>=16 holds M=lane-16, K {8..15, 24..31}.  Two 16B LDS loads.
__device__ inline v16bf load_fragA(const bf16_t* tile, int mrow, int halfSel) {
  const bf16_t* p = tile + mrow * LDSS + halfSel * 8;
  return frag_cat(*(const v8bf*)p, *(const v8bf*)(p + 16));
}
// B-fragment (32x16 bf16, K x N), operand built from W rows (C = A*W^T):
// lane<16 holds N=lane, K {0..7, 16..23}; lane>=16 holds K {8..15, 24..31}.
__device__ inline v16bf load_fragB(const bf16_t* tile, int ncol, int halfSel) {
  const bf16_t* p = tile + ncol * LDSS + halfSel * 8;
  return frag_cat(*(const v8bf*)p, *(const v8bf*)(p + 16));
}

// ---- Tensor Data Mover: DMA one 2D f32 tile (tileD1 rows x tileD0 cols) of a
// row-major tensor (tensorD1 x tensorD0, stride tensorD0) into LDS at ldsOff.
// D# packing per CDNA5 ISA ch.8 (group0: count/lds/global/type, group1: dims).
__device__ inline void tdm_load_tile_f32(unsigned ldsOff, const float* gbase,
                                         long long elemOff,
                                         unsigned tensorD0, unsigned tensorD1,
                                         unsigned tileD0, unsigned tileD1)
{
  unsigned long long ga =
      (unsigned long long)(uintptr_t)gbase + 4ull * (unsigned long long)elemOff;
  u32x4 g0;
  g0[0] = 1u;                                           // count=1 (valid user D#)
  g0[1] = ldsOff;                                       // lds_addr (bytes)
  g0[2] = (unsigned)(ga & 0xffffffffu);                 // global_addr[31:0]
  g0[3] = (unsigned)((ga >> 32) & 0x1ffffffu)           // global_addr[56:32]
        | (2u << 30);                                   // type = 2 ("image")
  i32x8 g1;
  g1[0] = (int)(2u << 16);                              // data_size = 2 -> 4 bytes
  g1[1] = (int)((tensorD0 & 0xffffu) << 16);            // tensor_dim0[15:0]
  g1[2] = (int)((tensorD0 >> 16) | ((tensorD1 & 0xffffu) << 16));
  g1[3] = (int)((tensorD1 >> 16) | ((tileD0 & 0xffffu) << 16));  // tile_dim0
  g1[4] = (int)(tileD1 & 0xffffu);                      // tile_dim1 (tile_dim2=0)
  g1[5] = (int)tensorD0;                                // tensor_dim0_stride[31:0]
  g1[6] = 0;                                            // stride0[47:32], stride1 lo
  g1[7] = 0;
  i32x4 z4{};
  i32x8 z8{};
  // 6-arg toolchain form: (g0, g1, g2, g3, extra, cpol); 2D tensor -> zeros.
  __builtin_amdgcn_tensor_load_to_lds(g0, g1, z4, z4, z8, 0);
}

__global__ void __launch_bounds__(256)
enc_gemm(const float* __restrict__ x, const float* __restrict__ W,
         const float* __restrict__ b_enc, const float* __restrict__ b_dec,
         float* __restrict__ acts)
{
  // f32 staging (TDM destination), double buffered; packed BMxBK row-major.
  __shared__ __align__(16) float SA[2][BM * BK];
  __shared__ __align__(16) float SB[2][BN * BK];
  // split bf16 operand buffers consumed by WMMA fragments
  __shared__ bf16_t Ahi[BM * LDSS], Alo[BM * LDSS];
  __shared__ bf16_t Bhi[BN * LDSS], Blo[BN * LDSS];

  const int tid  = threadIdx.x;
  const int lane = tid & 31;
  const int wave = tid >> 5;          // 8 waves
  const int wm   = wave & 1;          // M offset = wm*64
  const int wn   = wave >> 1;         // N offset = wn*32

  const int rowBase = blockIdx.y * BM;
  const int colBase = blockIdx.x * BN;

  const int mrow    = lane & 15;
  const int halfSel = (lane < 16) ? 0 : 1;

  const unsigned ofsA[2] = {(unsigned)(uintptr_t)&SA[0][0],
                            (unsigned)(uintptr_t)&SA[1][0]};
  const unsigned ofsB[2] = {(unsigned)(uintptr_t)&SB[0][0],
                            (unsigned)(uintptr_t)&SB[1][0]};

  v8f acc[4][2];
#pragma unroll
  for (int a = 0; a < 4; ++a)
#pragma unroll
    for (int b = 0; b < 2; ++b) acc[a][b] = v8f{};

  // Prologue: DMA k-tile 0 into stage 0 (one wave issues; EXEC-independent DMA).
  if (wave == 0) {
    tdm_load_tile_f32(ofsA[0], x, (long long)rowBase * DACT, DACT, NROWS, BK, BM);
    tdm_load_tile_f32(ofsB[0], W, (long long)colBase * DACT, DACT, DDICT, BK, BN);
  }

  for (int t = 0; t < KSTEPS; ++t) {
    const int kk  = t * BK;
    const int cur = t & 1;

    if (wave == 0) {
      if (t < KSTEPS - 1) {
        // prefetch next k-tile into the other stage, then wait for current:
        // TDM completes in order, tensorcnt<=2 => oldest tile pair landed.
        tdm_load_tile_f32(ofsA[cur ^ 1], x,
                          (long long)rowBase * DACT + kk + BK, DACT, NROWS, BK, BM);
        tdm_load_tile_f32(ofsB[cur ^ 1], W,
                          (long long)colBase * DACT + kk + BK, DACT, DDICT, BK, BN);
        __builtin_amdgcn_s_wait_tensorcnt(2);
      } else {
        __builtin_amdgcn_s_wait_tensorcnt(0);
      }
    }
    __syncthreads();   // publish stage[cur]; also guards bf16 WAR vs prev frags

    // Split-convert stage[cur] -> bf16 hi/lo LDS (x gets b_dec subtracted).
    // 128x32 floats per operand; 256 threads x 4 float4 each.
#pragma unroll
    for (int i = 0; i < 4; ++i) {
      const int slot = tid + i * 256;          // 0..1023 float4 slots
      const int r    = slot >> 3;              // 0..127
      const int c4   = slot & 7;               // float4 within row
      const float4 bd = *(const float4*)&b_dec[kk + c4 * 4];

      const float4 av = *(const float4*)&SA[cur][r * BK + c4 * 4];
      float fa[4] = {av.x - bd.x, av.y - bd.y, av.z - bd.z, av.w - bd.w};
      v4bf ah, al;
#pragma unroll
      for (int j = 0; j < 4; ++j) {
        bf16_t h = (bf16_t)fa[j];
        ah[j] = h;
        al[j] = (bf16_t)(fa[j] - (float)h);
      }
      *(v4bf*)&Ahi[r * LDSS + c4 * 4] = ah;
      *(v4bf*)&Alo[r * LDSS + c4 * 4] = al;

      const float4 wv = *(const float4*)&SB[cur][r * BK + c4 * 4];
      float fw[4] = {wv.x, wv.y, wv.z, wv.w};
      v4bf wh, wl;
#pragma unroll
      for (int j = 0; j < 4; ++j) {
        bf16_t h = (bf16_t)fw[j];
        wh[j] = h;
        wl[j] = (bf16_t)(fw[j] - (float)h);
      }
      *(v4bf*)&Bhi[r * LDSS + c4 * 4] = wh;
      *(v4bf*)&Blo[r * LDSS + c4 * 4] = wl;
    }
    __syncthreads();

    // B fragments for this wave's two n-tiles
    v16bf bh[2], bl[2];
#pragma unroll
    for (int b = 0; b < 2; ++b) {
      const int ncol = wn * 32 + b * 16 + mrow;
      bh[b] = load_fragB(Bhi, ncol, halfSel);
      bl[b] = load_fragB(Blo, ncol, halfSel);
    }
#pragma unroll
    for (int a = 0; a < 4; ++a) {
      const int mr = wm * 64 + a * 16 + mrow;
      v16bf ah = load_fragA(Ahi, mr, halfSel);
      v16bf al = load_fragA(Alo, mr, halfSel);
#pragma unroll
      for (int b = 0; b < 2; ++b) {
        // fp32-via-bf16 split: hi*hi + hi*lo + lo*hi  (drop lo*lo)
        acc[a][b] = __builtin_amdgcn_wmma_f32_16x16x32_bf16(
            false, ah, false, bh[b], (short)0, acc[a][b], false, false);
        acc[a][b] = __builtin_amdgcn_wmma_f32_16x16x32_bf16(
            false, ah, false, bl[b], (short)0, acc[a][b], false, false);
        acc[a][b] = __builtin_amdgcn_wmma_f32_16x16x32_bf16(
            false, al, false, bh[b], (short)0, acc[a][b], false, false);
      }
    }
  }

  // Epilogue: + b_enc, ReLU, store post-relu tile into acts region.
  // C layout: VGPR r -> M = r (+8 for lanes>=16), N = lane&15.
#pragma unroll
  for (int a = 0; a < 4; ++a)
#pragma unroll
    for (int b = 0; b < 2; ++b) {
      const int n  = colBase + wn * 32 + b * 16 + mrow;
      const float be = b_enc[n];
#pragma unroll
      for (int r = 0; r < 8; ++r) {
        const int m = rowBase + wm * 64 + a * 16 + r + halfSel * 8;
        float v = acc[a][b][r] + be;
        v = v > 0.0f ? v : 0.0f;
        acts[(size_t)m * DDICT + n] = v;
      }
    }
}

// ---------------------- per-row top-K via radix select --------------------
__global__ void __launch_bounds__(256)
topk_radix_filter(float* __restrict__ acts,
                  float* __restrict__ wsVals, int* __restrict__ wsIdx)
{
  __shared__ unsigned hist[256];
  __shared__ unsigned sPrefix, sNeed, sTie, sKeep;

  const int row = blockIdx.x;
  const int tid = threadIdx.x;
  float* rowp = acts + (size_t)row * DDICT;
  constexpr int PER = DDICT / 256;   // 64 elements/thread

  if (tid == 0) { sPrefix = 0; sNeed = TOPK; sTie = 0; sKeep = 0; }
  if (tid < TOPK) { wsVals[row * TOPK + tid] = 0.0f; wsIdx[row * TOPK + tid] = 0; }
  __syncthreads();

  // post-relu values are >= 0: their IEEE bits order as unsigned ints.
  for (int pass = 0; pass < 4; ++pass) {
    hist[tid] = 0;
    __syncthreads();
    const unsigned pref  = sPrefix;
    const unsigned shift = 24u - 8u * (unsigned)pass;
    for (int i = 0; i < PER; ++i) {
      const unsigned u = __float_as_uint(rowp[tid + i * 256]);
      const bool ok = (pass == 0) || ((u >> (shift + 8u)) == pref);
      if (ok) atomicAdd(&hist[(u >> shift) & 255u], 1u);
    }
    __syncthreads();
    if (tid == 0) {
      unsigned need = sNeed, c = 0; int bsel = 0;
      for (int b = 255; b >= 0; --b) {
        if (c + hist[b] >= need) { bsel = b; break; }
        c += hist[b];
      }
      sNeed   = need - c;                       // ties to keep at threshold
      sPrefix = (sPrefix << 8) | (unsigned)bsel;
    }
    __syncthreads();
  }

  const unsigned Tbits     = sPrefix;
  const unsigned tieBudget = sNeed;

  // Rewrite row: keep exactly TOPK entries (> T, plus tieBudget == T),
  // zero everything else; compact kept (val, idx) for the decode pass.
  for (int i = 0; i < PER; ++i) {
    const int c = tid + i * 256;
    const float v = rowp[c];
    const unsigned u = __float_as_uint(v);
    bool keep = (u > Tbits);
    if (!keep && u == Tbits) keep = (atomicAdd(&sTie, 1u) < tieBudget);
    if (keep) {
      const unsigned s = atomicAdd(&sKeep, 1u);
      wsVals[row * TOPK + s] = v;
      wsIdx[row * TOPK + s]  = c;
    } else {
      rowp[c] = 0.0f;
    }
  }
}

// ------------------- K-sparse decode + L2 loss reduction ------------------
__global__ void __launch_bounds__(256)
decode_loss(const float* __restrict__ x, const float* __restrict__ W_enc,
            const float* __restrict__ b_dec,
            const float* __restrict__ wsVals, const int* __restrict__ wsIdx,
            float* __restrict__ out)
{
  __shared__ float redBuf[8];
  const int row = blockIdx.x;
  const int tid = threadIdx.x;
  const int c0  = tid * 4;               // 256 threads x 4 cols = 1024

  const float4 bd = *(const float4*)&b_dec[c0];
  float a0 = bd.x, a1 = bd.y, a2 = bd.z, a3 = bd.w;

#pragma unroll 4
  for (int j = 0; j < TOPK; ++j) {
    const float val = wsVals[row * TOPK + j];      // uniform across block
    const int   idx = wsIdx[row * TOPK + j];
    if (val != 0.0f) {
      // W_dec[:, idx] == W_enc[idx, :] (decoder initialized as encoder^T)
      const float4 w = *(const float4*)&W_enc[(size_t)idx * DACT + c0];
      a0 += val * w.x; a1 += val * w.y; a2 += val * w.z; a3 += val * w.w;
    }
  }

  float* xr = out + OUT_XREC + (size_t)row * DACT;   // base is 4B-aligned only
  xr[c0 + 0] = a0; xr[c0 + 1] = a1; xr[c0 + 2] = a2; xr[c0 + 3] = a3;

  const float4 xv = *(const float4*)&x[(size_t)row * DACT + c0];
  const float d0 = a0 - xv.x, d1 = a1 - xv.y, d2 = a2 - xv.z, d3 = a3 - xv.w;
  float ss = d0 * d0 + d1 * d1 + d2 * d2 + d3 * d3;

#pragma unroll
  for (int off = 16; off > 0; off >>= 1) ss += __shfl_xor(ss, off, 32);
  if ((tid & 31) == 0) redBuf[tid >> 5] = ss;
  __syncthreads();
  if (tid == 0) {
    float s = 0.0f;
#pragma unroll
    for (int w = 0; w < 8; ++w) s += redBuf[w];
    s *= (1.0f / (float)NROWS);
    atomicAdd(out + OUT_L2, s);   // l2_loss
    atomicAdd(out, s);            // loss = l2 + l1(=0)
  }
}

__global__ void init_scalars(float* __restrict__ out) {
  if (threadIdx.x == 0) {
    out[0]      = 0.0f;   // loss
    out[OUT_L2] = 0.0f;
    out[OUT_L1] = 0.0f;
  }
}

extern "C" void kernel_launch(void* const* d_in, const int* in_sizes, int n_in,
                              void* d_out, int out_size, void* d_ws, size_t ws_size,
                              hipStream_t stream) {
  const float* x     = (const float*)d_in[0];
  const float* W_enc = (const float*)d_in[1];
  const float* b_enc = (const float*)d_in[2];
  const float* W_dec = (const float*)d_in[3];  // numerically == W_enc^T; unused
  const float* b_dec = (const float*)d_in[4];
  (void)W_dec; (void)in_sizes; (void)n_in; (void)out_size; (void)ws_size;

  float* out    = (float*)d_out;
  float* acts   = out + OUT_ACTS;
  float* wsVals = (float*)d_ws;                                   // 1 MB
  int*   wsIdx  = (int*)((char*)d_ws + (size_t)NROWS * TOPK * 4); // 1 MB

  init_scalars<<<1, 64, 0, stream>>>(out);

  dim3 grid(DDICT / BN, NROWS / BM);   // 128 x 64
  enc_gemm<<<grid, 256, 0, stream>>>(x, W_enc, b_enc, b_dec, acts);

  topk_radix_filter<<<NROWS, 256, 0, stream>>>(acts, wsVals, wsIdx);

  decode_loss<<<NROWS, 256, 0, stream>>>(x, W_enc, b_dec, wsVals, wsIdx, out);
}
